// DownsampleTransformerBlock_70016556860009
// MI455X (gfx1250) — matrix operand order repaired and checked by reading, not compiled, not run on hardware
//
#include <hip/hip_runtime.h>
#include <hip/hip_bf16.h>

typedef __bf16 bf16;
typedef __attribute__((ext_vector_type(16))) __bf16 v16bf;
typedef __attribute__((ext_vector_type(8)))  __bf16 v8bf;
typedef __attribute__((ext_vector_type(8)))  float  v8f;

#define BB 4
#define SS 2048
#define DD 256
#define HH 8
#define DKK 32
#define FF 1024
#define MM (BB*SS)

// ---------------- WMMA fragment helpers (gfx1250 wave32 layouts) ----------------
// A (16x32 bf16 row-major): lane l holds row M=l; element e -> K=e+hi*16
// => two contiguous 16B chunks.
__device__ __forceinline__ v16bf load_a16x32(const bf16* __restrict__ row, int hi) {
  const v8bf lo  = *(const v8bf*)(row + (hi ? 16 : 0));
  const v8bf hic = *(const v8bf*)(row + 8 + (hi ? 16 : 0));
  v16bf a;
  #pragma unroll
  for (int j = 0; j < 8; ++j) { a[j] = lo[j]; a[8 + j] = hic[j]; }
  return a;
}

// ---------------- weight pack: fp32 [K][N] row-major -> per-lane fragment order ----
// packed[(((nt*(K/32))+kt)*32 + lane)*16 + e] = W[kt*32 + (lane>>4)*16 + e][nt*16 + (lane&15)]
__global__ __launch_bounds__(256) void pack_b_kernel(const float* __restrict__ W,
                                                     bf16* __restrict__ P,
                                                     int Kt, int Nt) {
  int id = blockIdx.x * 256 + threadIdx.x;
  if (id >= Kt * Nt) return;
  int e = id & 15;
  int lane = (id >> 4) & 31;
  int rest = id >> 9;
  int ktiles = Kt >> 5;
  int kt = rest % ktiles, nt = rest / ktiles;
  int k = kt * 32 + (lane >> 4) * 16 + e;
  int n = nt * 16 + (lane & 15);
  P[id] = (bf16)W[(size_t)k * Nt + n];
}

__global__ __launch_bounds__(256) void zero_kernel(float* __restrict__ p, int n) {
  int i = blockIdx.x * 256 + threadIdx.x;
  if (i < n) p[i] = 0.0f;
}

// ---------------- LayerNorm (one wave per row of 256) ----------------
__global__ __launch_bounds__(256) void ln_kernel(const float* __restrict__ X,
                                                 const float* __restrict__ g,
                                                 const float* __restrict__ bta,
                                                 bf16* __restrict__ out,
                                                 bf16* __restrict__ xcopy, int Mrows) {
  int row = blockIdx.x * 8 + (threadIdx.x >> 5);
  int lane = threadIdx.x & 31;
  if (row >= Mrows) return;
  const float* xr = X + (size_t)row * DD;
  float vals[8];
  float s = 0.f, s2 = 0.f;
  #pragma unroll
  for (int i = 0; i < 8; ++i) {
    float v = xr[lane + 32 * i];
    vals[i] = v; s += v; s2 += v * v;
  }
  #pragma unroll
  for (int o = 16; o > 0; o >>= 1) {
    s  += __shfl_xor(s,  o, 32);
    s2 += __shfl_xor(s2, o, 32);
  }
  float mean = s * (1.0f / DD);
  float var  = s2 * (1.0f / DD) - mean * mean;
  float inv  = rsqrtf(var + 1e-6f);
  #pragma unroll
  for (int i = 0; i < 8; ++i) {
    int c = lane + 32 * i;
    out[(size_t)row * DD + c] = (bf16)((vals[i] - mean) * inv * g[c] + bta[c]);
    if (xcopy) xcopy[(size_t)row * DD + c] = (bf16)vals[i];
  }
}

// ---------------- generic WMMA GEMM over 16x64 strips, packed-B ----------------
__global__ __launch_bounds__(256) void gemm_kernel(const bf16* __restrict__ A,
                                                   const bf16* __restrict__ Bp,
                                                   const float* __restrict__ bias,
                                                   const float* __restrict__ residual,
                                                   float* __restrict__ outF,
                                                   bf16* __restrict__ outH,
                                                   int Mt, int Nt, int Kt, int relu) {
  int tid = threadIdx.x;
  int wave = tid >> 5, lane = tid & 31;
  int l = lane & 15, hi = lane >> 4;
  int tiles_m = Mt >> 4, strips_n = Nt >> 6;
  int t = blockIdx.x * 8 + wave;
  if (t >= tiles_m * strips_n) return;
  int tm = t % tiles_m, ts = t / tiles_m;
  int m0 = tm << 4, n0 = ts << 6;
  int ktiles = Kt >> 5;
  v8f acc0 = {}, acc1 = {}, acc2 = {}, acc3 = {};
  const bf16* arow = A + (size_t)(m0 + l) * Kt;
  for (int kt = 0; kt < ktiles; ++kt) {
    __builtin_prefetch(arow + kt * 32 + 128, 0, 1);
    v16bf a = load_a16x32(arow + kt * 32, hi);
    const v16bf* bp0 = (const v16bf*)(Bp + ((((size_t)(ts * 4 + 0) * ktiles) + kt) * 32 + lane) * 16);
    const v16bf* bp1 = (const v16bf*)(Bp + ((((size_t)(ts * 4 + 1) * ktiles) + kt) * 32 + lane) * 16);
    const v16bf* bp2 = (const v16bf*)(Bp + ((((size_t)(ts * 4 + 2) * ktiles) + kt) * 32 + lane) * 16);
    const v16bf* bp3 = (const v16bf*)(Bp + ((((size_t)(ts * 4 + 3) * ktiles) + kt) * 32 + lane) * 16);
    acc0 = __builtin_amdgcn_wmma_f32_16x16x32_bf16(false, a, false, *bp0, (short)0, acc0, false, false);
    acc1 = __builtin_amdgcn_wmma_f32_16x16x32_bf16(false, a, false, *bp1, (short)0, acc1, false, false);
    acc2 = __builtin_amdgcn_wmma_f32_16x16x32_bf16(false, a, false, *bp2, (short)0, acc2, false, false);
    acc3 = __builtin_amdgcn_wmma_f32_16x16x32_bf16(false, a, false, *bp3, (short)0, acc3, false, false);
  }
  v8f accs[4] = {acc0, acc1, acc2, acc3};
  #pragma unroll
  for (int q = 0; q < 4; ++q) {
    #pragma unroll
    for (int r = 0; r < 8; ++r) {
      int m = m0 + r + hi * 8, n = n0 + q * 16 + l;
      float v = accs[q][r];
      if (bias) v += bias[n];
      if (relu) v = fmaxf(v, 0.0f);
      if (residual) v += residual[(size_t)m * Nt + n];
      if (outF) outF[(size_t)m * Nt + n] = v;
      if (outH) outH[(size_t)m * Nt + n] = (bf16)v;
    }
  }
}

// ---------------- QKV projection: [M,256]@[256,256] -> Q/K [B,H,S,32] or V^T [B,H,32,S]
__global__ __launch_bounds__(256) void qkv_kernel(const bf16* __restrict__ A,
                                                  const bf16* __restrict__ Bp,
                                                  const float* __restrict__ bias,
                                                  bf16* __restrict__ out, int vT) {
  int tid = threadIdx.x;
  int wave = tid >> 5, lane = tid & 31;
  int l = lane & 15, hi = lane >> 4;
  const int tiles_m = MM >> 4;           // 512
  int t = blockIdx.x * 8 + wave;         // 2048 strip-tasks
  int tm = t % tiles_m, ts = t / tiles_m;
  int m0 = tm << 4, n0 = ts << 6;
  const int ktiles = DD >> 5;            // 8
  v8f acc0 = {}, acc1 = {}, acc2 = {}, acc3 = {};
  const bf16* arow = A + (size_t)(m0 + l) * DD;
  for (int kt = 0; kt < ktiles; ++kt) {
    v16bf a = load_a16x32(arow + kt * 32, hi);
    const v16bf* bp0 = (const v16bf*)(Bp + ((((size_t)(ts * 4 + 0) * ktiles) + kt) * 32 + lane) * 16);
    const v16bf* bp1 = (const v16bf*)(Bp + ((((size_t)(ts * 4 + 1) * ktiles) + kt) * 32 + lane) * 16);
    const v16bf* bp2 = (const v16bf*)(Bp + ((((size_t)(ts * 4 + 2) * ktiles) + kt) * 32 + lane) * 16);
    const v16bf* bp3 = (const v16bf*)(Bp + ((((size_t)(ts * 4 + 3) * ktiles) + kt) * 32 + lane) * 16);
    acc0 = __builtin_amdgcn_wmma_f32_16x16x32_bf16(false, a, false, *bp0, (short)0, acc0, false, false);
    acc1 = __builtin_amdgcn_wmma_f32_16x16x32_bf16(false, a, false, *bp1, (short)0, acc1, false, false);
    acc2 = __builtin_amdgcn_wmma_f32_16x16x32_bf16(false, a, false, *bp2, (short)0, acc2, false, false);
    acc3 = __builtin_amdgcn_wmma_f32_16x16x32_bf16(false, a, false, *bp3, (short)0, acc3, false, false);
  }
  v8f accs[4] = {acc0, acc1, acc2, acc3};
  #pragma unroll
  for (int q = 0; q < 4; ++q) {
    #pragma unroll
    for (int r = 0; r < 8; ++r) {
      int m = m0 + r + hi * 8, n = n0 + q * 16 + l;
      int b_ = m / SS, s_ = m % SS, h_ = n >> 5, kk = n & 31;
      float v = accs[q][r] + bias[n];
      if (vT)
        out[(((size_t)b_ * HH + h_) * DKK + kk) * SS + s_] = (bf16)v;  // [B,H,dk,S]
      else
        out[(((size_t)b_ * HH + h_) * SS + s_) * DKK + kk] = (bf16)v;  // [B,H,S,dk]
    }
  }
}

// ---------------- fused attention for one (b,h,16 query rows) ----------------
// scores (16x2048) in LDS as bf16 (64KB), softmax, importance column sums, P@V.
__global__ __launch_bounds__(256) void attn_kernel(const bf16* __restrict__ Q,
                                                   const bf16* __restrict__ Km,
                                                   const bf16* __restrict__ Vt,  // [B,H,dk,S]
                                                   bf16* __restrict__ O,    // [B,S,H*dk]
                                                   float* __restrict__ imp, // [B,S]
                                                   float wscale) {
  __shared__ bf16 sc[16][SS];            // exactly 64KB
  int tid = threadIdx.x;
  int wave = tid >> 5, lane = tid & 31;
  int l = lane & 15, hi = lane >> 4;

  int blk = blockIdx.x;                  // B*H*(S/16)
  int qb = blk % (SS / 16);
  int bh = blk / (SS / 16);
  int h = bh % HH, b = bh / HH;
  int q0 = qb * 16;

  const bf16* Qbh = Q + (size_t)bh * SS * DKK;
  const bf16* Kbh = Km + (size_t)bh * SS * DKK;
  const bf16* Vbh = Vt + (size_t)bh * DKK * SS;

  // Q fragment (16 rows x K=32), loaded once
  v16bf aq = load_a16x32(Qbh + (size_t)(q0 + l) * DKK, hi);

  const float scale = 0.17677669529663687f;  // 1/sqrt(32)

  // ---- scores: S = Q @ K^T * scale ; B[k][n]=K[n][k]: 16 contiguous bf16 per lane
  for (int n0 = wave * 16; n0 < SS; n0 += 8 * 16) {
    v16bf bk_frag = *(const v16bf*)(Kbh + (size_t)(n0 + l) * DKK + (hi ? 16 : 0));
    v8f c = {};
    c = __builtin_amdgcn_wmma_f32_16x16x32_bf16(false, aq, false, bk_frag, (short)0, c,
                                                false, false);
    #pragma unroll
    for (int r = 0; r < 8; ++r)
      sc[r + hi * 8][n0 + l] = (bf16)(c[r] * scale);
  }
  __syncthreads();

  // ---- softmax per row: 16 threads / row, shuffle reductions within 16-lane groups
  {
    int r = tid >> 4, t16 = tid & 15;
    float mx = -1e30f;
    for (int j = t16; j < SS; j += 16) mx = fmaxf(mx, (float)sc[r][j]);
    #pragma unroll
    for (int o = 8; o > 0; o >>= 1) mx = fmaxf(mx, __shfl_xor(mx, o, 32));
    float sum = 0.f;
    for (int j = t16; j < SS; j += 16) {
      float p = __expf((float)sc[r][j] - mx);
      sum += p;
      sc[r][j] = (bf16)p;
    }
    #pragma unroll
    for (int o = 8; o > 0; o >>= 1) sum += __shfl_xor(sum, o, 32);
    float inv = 1.0f / sum;
    for (int j = t16; j < SS; j += 16)
      sc[r][j] = (bf16)((float)sc[r][j] * inv);
  }
  __syncthreads();

  // ---- importance: column sums over the 16 query rows
  for (int c0 = tid; c0 < SS; c0 += 256) {
    float s = 0.f;
    #pragma unroll
    for (int rr = 0; rr < 16; ++rr) s += (float)sc[rr][c0];
    atomicAdd(&imp[(size_t)b * SS + c0], s * wscale);
  }
  __syncthreads();

  // ---- O = P @ V : 2 n-tiles x 4 k-chunks across 8 waves, split-K partials
  int ntile = wave >> 2;                 // 0..1
  int kc = wave & 3;                     // 0..3
  v8f acc = {};
  const bf16* vrow = Vbh + (size_t)(ntile * 16 + l) * SS;   // V^T row = dk index
  for (int k0 = kc * 512; k0 < kc * 512 + 512; k0 += 32) {
    v16bf ap = load_a16x32(&sc[l][k0], hi);                          // from LDS
    v16bf bv = *(const v16bf*)(vrow + k0 + (hi ? 16 : 0));           // contiguous
    acc = __builtin_amdgcn_wmma_f32_16x16x32_bf16(false, ap, false, bv, (short)0, acc,
                                                  false, false);
  }
  __syncthreads();  // P no longer needed; reuse LDS as fp32 scratch
  float* scratch = (float*)&sc[0][0];
  #pragma unroll
  for (int rr = 0; rr < 8; ++rr) scratch[((size_t)wave * 32 + lane) * 8 + rr] = acc[rr];
  __syncthreads();

  // ---- reduce 4 split-K partials and store bf16 to [B,S,H*dk]
  {
    int lane2 = tid & 31, r2 = tid >> 5;  // 0..7
    #pragma unroll
    for (int nt = 0; nt < 2; ++nt) {
      float s = 0.f;
      #pragma unroll
      for (int kcc = 0; kcc < 4; ++kcc)
        s += scratch[((size_t)(nt * 4 + kcc) * 32 + lane2) * 8 + r2];
      int m = r2 + (lane2 >> 4) * 8;
      int n = nt * 16 + (lane2 & 15);
      O[(size_t)(b * SS + q0 + m) * DD + h * DKK + n] = (bf16)s;
    }
  }
}

// ---------------- per-batch top-K selection (bitonic sorts in LDS) ----------------
__global__ __launch_bounds__(1024) void topk_kernel(const float* __restrict__ imp,
                                                    int* __restrict__ idx_out, int Kc) {
  __shared__ float val[SS];
  __shared__ int ind[SS];
  int b = blockIdx.x, tid = threadIdx.x;
  for (int i = tid; i < SS; i += 1024) { val[i] = imp[(size_t)b * SS + i]; ind[i] = i; }
  __syncthreads();
  // bitonic sort DESCENDING by value
  for (int k = 2; k <= SS; k <<= 1) {
    for (int j = k >> 1; j > 0; j >>= 1) {
      #pragma unroll 1
      for (int w = 0; w < 2; ++w) {
        int i = tid + w * 1024;
        int ixj = i ^ j;
        if (ixj > i) {
          bool up = ((i & k) == 0);
          float vi = val[i], vx = val[ixj];
          bool sw = up ? (vi < vx) : (vi > vx);
          if (sw) {
            val[i] = vx; val[ixj] = vi;
            int t = ind[i]; ind[i] = ind[ixj]; ind[ixj] = t;
          }
        }
      }
      __syncthreads();
    }
  }
  // keep first Kc, sort those indices ASCENDING (sentinel-pad the rest)
  for (int i = tid; i < SS; i += 1024)
    if (i >= Kc) ind[i] = 0x7fffffff;
  __syncthreads();
  for (int k = 2; k <= SS; k <<= 1) {
    for (int j = k >> 1; j > 0; j >>= 1) {
      #pragma unroll 1
      for (int w = 0; w < 2; ++w) {
        int i = tid + w * 1024;
        int ixj = i ^ j;
        if (ixj > i) {
          bool up = ((i & k) == 0);
          int vi = ind[i], vx = ind[ixj];
          bool sw = up ? (vi > vx) : (vi < vx);
          if (sw) { ind[i] = vx; ind[ixj] = vi; }
        }
      }
      __syncthreads();
    }
  }
  for (int i = tid; i < Kc; i += 1024) idx_out[(size_t)b * Kc + i] = ind[i];
}

// ---------------- gather + residual proj add + BatchNorm -> d_out ----------------
__global__ __launch_bounds__(256) void gather_bn_kernel(const float* __restrict__ out3,
                                                        const float* __restrict__ xr,
                                                        const int* __restrict__ idx,
                                                        const float* __restrict__ bn_g,
                                                        const float* __restrict__ bn_b,
                                                        const float* __restrict__ bn_mean,
                                                        const float* __restrict__ bn_var,
                                                        float* __restrict__ down,
                                                        float* __restrict__ idx_f, int Kc) {
  int j = blockIdx.x;               // b*Kc + k
  int b = j / Kc;
  int s = idx[j];
  int d = threadIdx.x;
  size_t off = ((size_t)b * SS + s) * DD + d;
  float v = out3[off] + xr[off];
  v = (v - bn_mean[d]) * rsqrtf(bn_var[d] + 1e-3f) * bn_g[d] + bn_b[d];
  down[(size_t)j * DD + d] = v;
  if (d == 0) idx_f[j] = (float)s;
}

// ---------------- host driver ----------------
extern "C" void kernel_launch(void* const* d_in, const int* in_sizes, int n_in,
                              void* d_out, int out_size, void* d_ws, size_t ws_size,
                              hipStream_t stream) {
  (void)in_sizes; (void)n_in; (void)ws_size;
  const float* x      = (const float*)d_in[0];
  const float* ln1_g  = (const float*)d_in[1];
  const float* ln1_b  = (const float*)d_in[2];
  const float* ln2_g  = (const float*)d_in[3];
  const float* ln2_b  = (const float*)d_in[4];
  const float* ln3_g  = (const float*)d_in[5];
  const float* ln3_b  = (const float*)d_in[6];
  const float* gq_w = (const float*)d_in[7];  const float* gq_b = (const float*)d_in[8];
  const float* gk_w = (const float*)d_in[9];  const float* gk_b = (const float*)d_in[10];
  const float* gv_w = (const float*)d_in[11]; const float* gv_b = (const float*)d_in[12];
  const float* go_w = (const float*)d_in[13]; const float* go_b = (const float*)d_in[14];
  const float* lq_w = (const float*)d_in[15]; const float* lq_b = (const float*)d_in[16];
  const float* lk_w = (const float*)d_in[17]; const float* lk_b = (const float*)d_in[18];
  const float* lv_w = (const float*)d_in[19]; const float* lv_b = (const float*)d_in[20];
  const float* lo_w = (const float*)d_in[21]; const float* lo_b = (const float*)d_in[22];
  const float* ffn_w1 = (const float*)d_in[23]; const float* ffn_b1 = (const float*)d_in[24];
  const float* ffn_w2 = (const float*)d_in[25]; const float* ffn_b2 = (const float*)d_in[26];
  const float* res_w  = (const float*)d_in[27]; const float* res_b  = (const float*)d_in[28];
  const float* bn_g   = (const float*)d_in[29]; const float* bn_b   = (const float*)d_in[30];
  const float* bn_mean= (const float*)d_in[31]; const float* bn_var = (const float*)d_in[32];

  int Kc = (out_size > 0) ? out_size / (BB * (DD + 1)) : (SS / 10);
  if (Kc < 1) Kc = 1;
  if (Kc > SS) Kc = SS;

  // workspace carve-out
  char* wp = (char*)d_ws;
  auto carve = [&](size_t bytes) -> void* {
    void* p = (void*)wp;
    wp += (bytes + 255) & ~(size_t)255;
    return p;
  };
  bf16* wqg = (bf16*)carve(DD * DD * 2); bf16* wkg = (bf16*)carve(DD * DD * 2);
  bf16* wvg = (bf16*)carve(DD * DD * 2); bf16* wog = (bf16*)carve(DD * DD * 2);
  bf16* wql = (bf16*)carve(DD * DD * 2); bf16* wkl = (bf16*)carve(DD * DD * 2);
  bf16* wvl = (bf16*)carve(DD * DD * 2); bf16* wol = (bf16*)carve(DD * DD * 2);
  bf16* w1b = (bf16*)carve((size_t)DD * FF * 2);
  bf16* w2b = (bf16*)carve((size_t)FF * DD * 2);
  bf16* wrb = (bf16*)carve(DD * DD * 2);
  bf16* xb  = (bf16*)carve((size_t)MM * DD * 2);   // x in bf16
  bf16* nb  = (bf16*)carve((size_t)MM * DD * 2);   // current LN output, bf16
  bf16* qb  = (bf16*)carve((size_t)MM * DD * 2);   // [B,H,S,dk]
  bf16* kb  = (bf16*)carve((size_t)MM * DD * 2);   // [B,H,S,dk]
  bf16* vb  = (bf16*)carve((size_t)MM * DD * 2);   // [B,H,dk,S]  (transposed)
  bf16* aob = (bf16*)carve((size_t)MM * DD * 2);   // attention output [B,S,256]
  bf16* hb  = (bf16*)carve((size_t)MM * FF * 2);   // ffn hidden
  float* out1 = (float*)carve((size_t)MM * DD * 4);
  float* out2 = (float*)carve((size_t)MM * DD * 4);
  float* out3 = (float*)carve((size_t)MM * DD * 4);
  float* xr   = (float*)carve((size_t)MM * DD * 4);
  float* imp  = (float*)carve((size_t)BB * SS * 4);
  int*   idx  = (int*)carve((size_t)BB * SS * 4);

  auto pack = [&](const float* s, bf16* d, int K, int N) {
    pack_b_kernel<<<(K * N + 255) / 256, 256, 0, stream>>>(s, d, K, N);
  };
  pack(gq_w, wqg, DD, DD); pack(gk_w, wkg, DD, DD);
  pack(gv_w, wvg, DD, DD); pack(go_w, wog, DD, DD);
  pack(lq_w, wql, DD, DD); pack(lk_w, wkl, DD, DD);
  pack(lv_w, wvl, DD, DD); pack(lo_w, wol, DD, DD);
  pack(ffn_w1, w1b, DD, FF); pack(ffn_w2, w2b, FF, DD);
  pack(res_w, wrb, DD, DD);

  zero_kernel<<<(BB * SS + 255) / 256, 256, 0, stream>>>(imp, BB * SS);

  const int strip_blocks_256 = (MM / 16) * (DD / 64) / 8;   // 256
  const int strip_blocks_ffn1 = (MM / 16) * (FF / 64) / 8;  // 1024
  const int attn_blocks = BB * HH * (SS / 16);              // 4096

  // ---- global MHA ----
  ln_kernel<<<MM / 8, 256, 0, stream>>>(x, ln1_g, ln1_b, nb, xb, MM);
  qkv_kernel<<<strip_blocks_256, 256, 0, stream>>>(nb, wqg, gq_b, qb, 0);
  qkv_kernel<<<strip_blocks_256, 256, 0, stream>>>(nb, wkg, gk_b, kb, 0);
  qkv_kernel<<<strip_blocks_256, 256, 0, stream>>>(nb, wvg, gv_b, vb, 1);
  attn_kernel<<<attn_blocks, 256, 0, stream>>>(qb, kb, vb, aob, imp,
                                               0.7f / (float)(HH * SS));
  gemm_kernel<<<strip_blocks_256, 256, 0, stream>>>(aob, wog, go_b, x, out1, nullptr,
                                                    MM, DD, DD, 0);
  // ---- local MHA ----
  ln_kernel<<<MM / 8, 256, 0, stream>>>(out1, ln2_g, ln2_b, nb, nullptr, MM);
  qkv_kernel<<<strip_blocks_256, 256, 0, stream>>>(nb, wql, lq_b, qb, 0);
  qkv_kernel<<<strip_blocks_256, 256, 0, stream>>>(nb, wkl, lk_b, kb, 0);
  qkv_kernel<<<strip_blocks_256, 256, 0, stream>>>(nb, wvl, lv_b, vb, 1);
  attn_kernel<<<attn_blocks, 256, 0, stream>>>(qb, kb, vb, aob, imp,
                                               0.3f / (float)(HH * SS));
  gemm_kernel<<<strip_blocks_256, 256, 0, stream>>>(aob, wol, lo_b, out1, out2, nullptr,
                                                    MM, DD, DD, 0);
  // ---- FFN ----
  ln_kernel<<<MM / 8, 256, 0, stream>>>(out2, ln3_g, ln3_b, nb, nullptr, MM);
  gemm_kernel<<<strip_blocks_ffn1, 256, 0, stream>>>(nb, w1b, ffn_b1, nullptr,
                                                     nullptr, hb, MM, FF, DD, 1);
  gemm_kernel<<<strip_blocks_256, 256, 0, stream>>>(hb, w2b, ffn_b2, out2, out3, nullptr,
                                                    MM, DD, FF, 0);
  // ---- residual projection ----
  gemm_kernel<<<strip_blocks_256, 256, 0, stream>>>(xb, wrb, res_b, nullptr, xr, nullptr,
                                                    MM, DD, DD, 0);
  // ---- top-k + gather + BN ----
  topk_kernel<<<BB, 1024, 0, stream>>>(imp, idx, Kc);
  float* down  = (float*)d_out;
  float* idx_f = down + (size_t)BB * Kc * DD;
  gather_bn_kernel<<<BB * Kc, 256, 0, stream>>>(out3, xr, idx, bn_g, bn_b, bn_mean,
                                                bn_var, down, idx_f, Kc);
}
